// Attention_84628035600788
// MI455X (gfx1250) — matrix operand order repaired and checked by reading, not compiled, not run on hardware
//
#include <hip/hip_runtime.h>
#include <hip/hip_bf16.h>

// ---------------------------------------------------------------------------
// LoRA-fused multi-head attention for MI455X (gfx1250, wave32, WMMA bf16)
//
//   W_eff = W + (1/16) * B @ A           (precomputed, bf16)
//   q = xb @ Wq_eff^T ; k = xb @ Wk_eff^T ; Vt = Wv_eff @ xb^T
//   flash attention (online softmax) with v_wmma_f32_16x16x32_bf16,
//   K/V tiles staged via async global->LDS DMA (double-buffered, ASYNCcnt)
//   out = ao @ Wo_eff^T + bo   (fp32)
// ---------------------------------------------------------------------------

#define DIM      1024
#define HEADS    16
#define HEAD_DIM 64
#define SEQ      1024
#define BATCH    8
#define ROWS     (BATCH * SEQ)        // 8192 tokens
#define SCALING  0.0625f              // 1/RANK
#define SCALE    0.125f               // HEAD_DIM^-0.5

typedef __attribute__((ext_vector_type(16))) __bf16 v16bf;
typedef __attribute__((ext_vector_type(8)))  float  v8f;

union Frag {
    v16bf bf;
    uint4 u4[2];
    unsigned short us[16];
};

__device__ __forceinline__ unsigned short f2bf(float f) {
    union { float f; unsigned int u; } c;
    c.f = f;
    unsigned int u = c.u;
    u += 0x7FFFu + ((u >> 16) & 1u);    // round to nearest even
    return (unsigned short)(u >> 16);
}

// Async global->LDS 16-byte copy (CDNA5, tracked by ASYNCcnt).
// vdst operand supplies the LDS byte address (flat shared addr truncates to it).
__device__ __forceinline__ void async_b128(const unsigned short* g, unsigned short* l) {
    unsigned loff = (unsigned)(size_t)l;
    unsigned long long ga = (unsigned long long)(size_t)g;
    asm volatile("global_load_async_to_lds_b128 %0, %1, off"
                 :: "v"(loff), "v"(ga) : "memory");
}

// --------------------------------------------------------------------------
// fp32 -> bf16 bulk convert
// --------------------------------------------------------------------------
__global__ __launch_bounds__(256) void cvt_bf16_kernel(const float* __restrict__ x,
                                                       unsigned short* __restrict__ o,
                                                       int ntot) {
    int i = blockIdx.x * 256 + threadIdx.x;
    if (i < ntot) o[i] = f2bf(x[i]);
}

// --------------------------------------------------------------------------
// W_eff[n,k] = W[n,k] + SCALING * sum_r Blo[n,r] * Alo[r,k]   -> bf16
// --------------------------------------------------------------------------
__global__ __launch_bounds__(256) void fuse_weight_kernel(const float* __restrict__ W,
                                                          const float* __restrict__ Alo,
                                                          const float* __restrict__ Blo,
                                                          unsigned short* __restrict__ out) {
    int idx = blockIdx.x * 256 + threadIdx.x;   // n * DIM + k
    int n = idx >> 10;
    int k = idx & (DIM - 1);
    float acc = W[idx];
#pragma unroll
    for (int r = 0; r < 16; ++r)
        acc += SCALING * Blo[n * 16 + r] * Alo[r * DIM + k];
    out[idx] = f2bf(acc);
}

// --------------------------------------------------------------------------
// Generic GEMM: C[M,N] = A[M,K] * B[N,K]^T   (A,B row-major bf16)
// mode 0: fp32 out + bias ; mode 1: bf16 out (row-major)
// block = 128 threads (4 waves); wave tile = 64x64; block tile = 128x128
// grid = (N/128, M/128)
// --------------------------------------------------------------------------
__global__ __launch_bounds__(128) void gemm_bf16_kernel(const unsigned short* __restrict__ A,
                                                        const unsigned short* __restrict__ Bw,
                                                        const float* __restrict__ bias,
                                                        float* __restrict__ outF,
                                                        unsigned short* __restrict__ outB,
                                                        int M, int N, int K, int mode) {
    const int lane = threadIdx.x & 31;
    const int w    = threadIdx.x >> 5;
    const int col  = lane & 15;
    const int half = lane >> 4;
    const int bm   = blockIdx.y * 128 + (w & 1) * 64;
    const int bn   = blockIdx.x * 128 + (w >> 1) * 64;

    v8f c[4][4] = {};

    for (int kk = 0; kk < K; kk += 32) {
        Frag a[4], bfr[4];
#pragma unroll
        for (int mi = 0; mi < 4; ++mi) {
            // A fragment 16x32: lane-half selects K chunks {0..7,16..23}/{8..15,24..31}
            const unsigned short* p = A + (size_t)(bm + mi * 16 + col) * K + kk + half * 8;
            a[mi].u4[0] = *(const uint4*)(p);
            a[mi].u4[1] = *(const uint4*)(p + 16);
        }
#pragma unroll
        for (int nj = 0; nj < 4; ++nj) {
            // B fragment 32x16: lane-half selects K chunks {0..7,16..23}/{8..15,24..31}
            const unsigned short* p = Bw + (size_t)(bn + nj * 16 + col) * K + kk + half * 8;
            bfr[nj].u4[0] = *(const uint4*)(p);
            bfr[nj].u4[1] = *(const uint4*)(p + 16);
        }
#pragma unroll
        for (int mi = 0; mi < 4; ++mi)
#pragma unroll
            for (int nj = 0; nj < 4; ++nj)
                c[mi][nj] = __builtin_amdgcn_wmma_f32_16x16x32_bf16(
                    false, a[mi].bf, false, bfr[nj].bf, (short)0, c[mi][nj], false, false);
    }

    // Epilogue: C layout rows m = v + 8*half (per VGPR), cols n = lane%16
#pragma unroll
    for (int mi = 0; mi < 4; ++mi)
#pragma unroll
        for (int nj = 0; nj < 4; ++nj) {
            const int n = bn + nj * 16 + col;
#pragma unroll
            for (int v = 0; v < 8; ++v) {
                const int m = bm + mi * 16 + v + half * 8;
                const float val = c[mi][nj][v];
                if (mode == 0)
                    outF[(size_t)m * N + n] = val + bias[n];
                else
                    outB[(size_t)m * N + n] = f2bf(val);
            }
        }
}

// --------------------------------------------------------------------------
// Flash attention: block 128 threads = 4 waves, each wave owns 16 query rows.
// grid = (SEQ/64, HEADS, BATCH)
// Q,K in bf16 row-major (token, DIM); Vt transposed bf16 (DIM, tokens).
// K/V 32-key tiles staged cooperatively into LDS with async DMA, double
// buffered: copy of tile i+1 overlaps compute of tile i.
// --------------------------------------------------------------------------
#define KTILE_E (32 * HEAD_DIM)     // 2048 elems = 4KB
#define VTILE_E (HEAD_DIM * 32)     // 2048 elems = 4KB

// Each thread t (0..127) copies 32B of K and 32B of V (4 async b128 ops/tile).
__device__ __forceinline__ void stage_kv(const unsigned short* Kg, const unsigned short* Vg,
                                         unsigned short* kl, unsigned short* vl, int t) {
    const int jl = t >> 2, dd = (t & 3) * 16;          // K: row j_local, 16-elem chunk
    const unsigned short* gk = Kg + (size_t)jl * DIM + dd;
    unsigned short* lk = kl + jl * HEAD_DIM + dd;
    async_b128(gk, lk);
    async_b128(gk + 8, lk + 8);
    const int dr = t >> 1, jo = (t & 1) * 16;          // V: row d, 16-key chunk
    const unsigned short* gv = Vg + (size_t)dr * ROWS + jo;
    unsigned short* lv = vl + dr * 32 + jo;
    async_b128(gv, lv);
    async_b128(gv + 8, lv + 8);
}

__global__ __launch_bounds__(128) void attn_kernel(const unsigned short* __restrict__ Q,
                                                   const unsigned short* __restrict__ Kb,
                                                   const unsigned short* __restrict__ Vt,
                                                   unsigned short* __restrict__ O) {
    __shared__ __align__(16) unsigned short kls[2 * KTILE_E];   // K tiles [j][d]
    __shared__ __align__(16) unsigned short vls[2 * VTILE_E];   // V tiles [d][j]
    __shared__ __align__(16) unsigned short pls[4 * 16 * 32];   // P staging, 1KB/wave

    const int t    = threadIdx.x;
    const int lane = t & 31;
    const int w    = t >> 5;
    const int col  = lane & 15;
    const int half = lane >> 4;
    const int b    = blockIdx.z;
    const int h    = blockIdx.y;
    const int q0   = blockIdx.x * 64 + w * 16;
    unsigned short* pl = pls + w * 512;

    const unsigned short* Kbase = Kb + (size_t)(b * SEQ) * DIM + h * HEAD_DIM;
    const unsigned short* Vbase = Vt + (size_t)(h * HEAD_DIM) * ROWS + b * SEQ;

    // Prime the async pipeline: tiles 0 and 1 in flight (8 ops/wave).
    stage_kv(Kbase, Vbase, kls, vls, t);
    stage_kv(Kbase + (size_t)32 * DIM, Vbase + 32, kls + KTILE_E, vls + VTILE_E, t);

    // Load Q fragments (16 rows x 64 d as two 16x32 A-fragments)
    Frag aq[2];
    {
        const unsigned short* qp = Q + (size_t)(b * SEQ + q0 + col) * DIM + h * HEAD_DIM;
#pragma unroll
        for (int dq = 0; dq < 2; ++dq) {
            aq[dq].u4[0] = *(const uint4*)(qp + dq * 32 + half * 8);
            aq[dq].u4[1] = *(const uint4*)(qp + dq * 32 + 16 + half * 8);
        }
    }

    v8f o[4] = {};
    float mrow[8], lrow[8];
#pragma unroll
    for (int v = 0; v < 8; ++v) { mrow[v] = -__builtin_huge_valf(); lrow[v] = 0.0f; }

    const int NTILES = SEQ / 32;                 // 32 key-tiles
    for (int ib = 0; ib < NTILES; ++ib) {
        // Retire tile ib (in-order completion); tile ib+1 may stay in flight.
        if (ib + 1 < NTILES)
            asm volatile("s_wait_asynccnt 0x4" ::: "memory");
        else
            asm volatile("s_wait_asynccnt 0x0" ::: "memory");
        __syncthreads();

        const unsigned short* kt = kls + (ib & 1) * KTILE_E;
        const unsigned short* vts = vls + (ib & 1) * VTILE_E;

        // ---- S = (Q K^T) * SCALE for 32 keys (two 16x16 tiles) ----
        v8f s[2];
#pragma unroll
        for (int jt = 0; jt < 2; ++jt) {
            v8f acc = {};
#pragma unroll
            for (int ks = 0; ks < 2; ++ks) {
                Frag bk;
                const unsigned short* kp = kt + (jt * 16 + col) * HEAD_DIM + ks * 32 + half * 8;
                bk.u4[0] = *(const uint4*)(kp);
                bk.u4[1] = *(const uint4*)(kp + 16);
                acc = __builtin_amdgcn_wmma_f32_16x16x32_bf16(
                    false, aq[ks].bf, false, bk.bf, (short)0, acc, false, false);
            }
            s[jt] = acc * SCALE;
        }

        // ---- online softmax update (row stats live at VGPR slot v, 16-lane group) ----
        float fac[8];
#pragma unroll
        for (int v = 0; v < 8; ++v) {
            float tmax = fmaxf(s[0][v], s[1][v]);
#pragma unroll
            for (int off = 1; off < 16; off <<= 1)
                tmax = fmaxf(tmax, __shfl_xor(tmax, off, 32));
            const float nm = fmaxf(mrow[v], tmax);
            fac[v] = __expf(mrow[v] - nm);
            mrow[v] = nm;
        }
#pragma unroll
        for (int v = 0; v < 8; ++v) {
            const float p0 = __expf(s[0][v] - mrow[v]);
            const float p1 = __expf(s[1][v] - mrow[v]);
            s[0][v] = p0; s[1][v] = p1;
            float rs = p0 + p1;
#pragma unroll
            for (int off = 1; off < 16; off <<= 1)
                rs += __shfl_xor(rs, off, 32);
            lrow[v] = lrow[v] * fac[v] + rs;
        }
#pragma unroll
        for (int dt = 0; dt < 4; ++dt)
#pragma unroll
            for (int v = 0; v < 8; ++v)
                o[dt][v] *= fac[v];

        // ---- P: C-layout -> A-layout transpose through LDS (bf16) ----
#pragma unroll
        for (int jt = 0; jt < 2; ++jt)
#pragma unroll
            for (int v = 0; v < 8; ++v) {
                const int row = v + half * 8;
                pl[row * 32 + jt * 16 + col] = f2bf(s[jt][v]);
            }
        Frag pf;
        {
            const unsigned short* pp = pl + (size_t)col * 32 + half * 8;
            pf.u4[0] = *(const uint4*)(pp);
            pf.u4[1] = *(const uint4*)(pp + 16);
        }

        // ---- O += P @ V  (V tile in LDS: [d][j], contiguous along keys) ----
#pragma unroll
        for (int dt = 0; dt < 4; ++dt) {
            Frag bv;
            const unsigned short* vp = vts + (dt * 16 + col) * 32 + half * 8;
            bv.u4[0] = *(const uint4*)(vp);
            bv.u4[1] = *(const uint4*)(vp + 16);
            o[dt] = __builtin_amdgcn_wmma_f32_16x16x32_bf16(
                false, pf.bf, false, bv.bf, (short)0, o[dt], false, false);
        }

        // All waves done reading buf (ib&1); refill it with tile ib+2.
        __syncthreads();
        if (ib + 2 < NTILES) {
            const int jn = (ib + 2) * 32;
            stage_kv(Kbase + (size_t)jn * DIM, Vbase + jn,
                     kls + (ib & 1) * KTILE_E, vls + (ib & 1) * VTILE_E, t);
        }
    }

    // ---- normalize and write bf16 ----
#pragma unroll
    for (int dt = 0; dt < 4; ++dt)
#pragma unroll
        for (int v = 0; v < 8; ++v) {
            const int row = q0 + v + half * 8;
            const float val = o[dt][v] / lrow[v];
            O[(size_t)(b * SEQ + row) * DIM + h * HEAD_DIM + dt * 16 + col] = f2bf(val);
        }
}

// --------------------------------------------------------------------------
extern "C" void kernel_launch(void* const* d_in, const int* in_sizes, int n_in,
                              void* d_out, int out_size, void* d_ws, size_t ws_size,
                              hipStream_t stream) {
    const float* x  = (const float*)d_in[0];
    const float* Wq = (const float*)d_in[1];
    const float* Wk = (const float*)d_in[2];
    const float* Wv = (const float*)d_in[3];
    const float* Wo = (const float*)d_in[4];
    const float* bo = (const float*)d_in[5];
    const float* Aq = (const float*)d_in[6];
    const float* Bq = (const float*)d_in[7];
    const float* Ak = (const float*)d_in[8];
    const float* Bk = (const float*)d_in[9];
    const float* Av = (const float*)d_in[10];
    const float* Bv = (const float*)d_in[11];
    const float* Ao = (const float*)d_in[12];
    const float* Bo = (const float*)d_in[13];

    const size_t MN = (size_t)ROWS * DIM;        // 8M elems
    const size_t WW = (size_t)DIM * DIM;         // 1M elems

    unsigned short* ws16 = (unsigned short*)d_ws;
    unsigned short* xb  = ws16;                  // x in bf16
    unsigned short* wqe = xb  + MN;
    unsigned short* wke = wqe + WW;
    unsigned short* wve = wke + WW;
    unsigned short* woe = wve + WW;
    unsigned short* qb  = woe + WW;
    unsigned short* kb  = qb  + MN;
    unsigned short* vt  = kb  + MN;              // transposed V (DIM x ROWS)
    unsigned short* ao  = vt  + MN;              // attention output bf16

    // 1) convert activations, fuse LoRA into weights (bf16)
    cvt_bf16_kernel<<<(int)(MN / 256), 256, 0, stream>>>(x, xb, (int)MN);
    fuse_weight_kernel<<<(int)(WW / 256), 256, 0, stream>>>(Wq, Aq, Bq, wqe);
    fuse_weight_kernel<<<(int)(WW / 256), 256, 0, stream>>>(Wk, Ak, Bk, wke);
    fuse_weight_kernel<<<(int)(WW / 256), 256, 0, stream>>>(Wv, Av, Bv, wve);
    fuse_weight_kernel<<<(int)(WW / 256), 256, 0, stream>>>(Wo, Ao, Bo, woe);

    // 2) Q, K projections: (8192x1024) = xb @ W^T
    dim3 gQK(DIM / 128, ROWS / 128);
    gemm_bf16_kernel<<<gQK, 128, 0, stream>>>(xb, wqe, nullptr, nullptr, qb,
                                              ROWS, DIM, DIM, 1);
    gemm_bf16_kernel<<<gQK, 128, 0, stream>>>(xb, wke, nullptr, nullptr, kb,
                                              ROWS, DIM, DIM, 1);
    // 3) transposed V: Vt(1024x8192) = Wv_eff @ xb^T (swap A/B roles)
    dim3 gV(ROWS / 128, DIM / 128);
    gemm_bf16_kernel<<<gV, 128, 0, stream>>>(wve, xb, nullptr, nullptr, vt,
                                             DIM, ROWS, DIM, 1);

    // 4) flash attention (async-DMA staged K/V)
    dim3 gA(SEQ / 64, HEADS, BATCH);
    attn_kernel<<<gA, 128, 0, stream>>>(qb, kb, vt, ao);

    // 5) output projection with bias, fp32 result
    gemm_bf16_kernel<<<gQK, 128, 0, stream>>>(ao, woe, bo, (float*)d_out, nullptr,
                                              ROWS, DIM, DIM, 0);
}
